// ConditionalRandomField_37778532335879
// MI455X (gfx1250) — matrix operand order repaired and checked by reading, not compiled, not run on hardware
//
#include <hip/hip_runtime.h>
#include <hip/hip_bf16.h>

// CRF log-likelihood for B=256, L=512, T=128 on gfx1250.
// Forward recursion recast as 511 sequential (256x128)@(128x128) GEMMs using
// v_wmma_f32_16x16x32_bf16 with 3-product bf16 error compensation.
// E = exp(transitions) lives permanently in VGPRs as WMMA B-fragments.
// Raw hw transcendentals (v_exp_f32/v_log_f32) — operands provably normal.

#define BB 256
#define LL 512
#define TT 128

#define LN2F    0.6931471805599453f
#define LOG2EF  1.4426950408889634f

typedef __attribute__((ext_vector_type(16))) __bf16 v16bf;
typedef __attribute__((ext_vector_type(8)))  float  v8f;

union AFrag { v16bf v; uint4 q[2]; };

__device__ __forceinline__ unsigned int bf16_pack(float a, float b) {
    __bf16 ha = (__bf16)a, hb = (__bf16)b;
    return (unsigned int)__builtin_bit_cast(unsigned short, ha) |
           ((unsigned int)__builtin_bit_cast(unsigned short, hb) << 16);
}
__device__ __forceinline__ float bf16_hi_val(float a) {
    return (float)(__bf16)a;
}

__global__ void __launch_bounds__(256)
crf_forward_kernel(const float* __restrict__ inputs,
                   const float* __restrict__ trans,
                   const float* __restrict__ start_t,
                   const float* __restrict__ end_t,
                   const int*   __restrict__ mask,
                   float* __restrict__ den)
{
    // padded strides: 132 floats (528B) and 136 halves (272B) => conflict-free
    // row gathers for the A-fragment b128 loads (16 lanes hit distinct banks).
    __shared__ __align__(16) float          s_alpha[16][132];
    __shared__ __align__(16) unsigned short s_pah[16][136];
    __shared__ __align__(16) unsigned short s_pal[16][136];
    __shared__ float s_rowmax[16];
    __shared__ int   s_len[16];

    const int tid  = threadIdx.x;
    const int wv   = tid >> 5;      // wave id 0..7 -> j tile
    const int ln   = tid & 31;
    const int jn   = ln & 15;
    const int hi16 = ln >> 4;       // 0: lanes 0-15, 1: lanes 16-31
    const int b0   = blockIdx.x * 16;
    const int j    = wv * 16 + jn;  // this lane's output column (tag j)

    // ---- persistent E = exp(transitions), bf16 hi/lo split, B-matrix layout
    // (32x16 bf16 B frag, contiguous k order used here: lanes 0-15 hold K=0..15, lanes 16-31 hold K=16..31,
    //  VGPR v packs K=2v (lo16) and K=2v+1 (hi16))
    v16bf ehi[4], elo[4];
    #pragma unroll
    for (int blk = 0; blk < 4; ++blk) {
        #pragma unroll
        for (int v = 0; v < 8; ++v) {
            int k = blk * 32 + (hi16 ? 16 : 0) + 2 * v;
            float e0 = __builtin_amdgcn_exp2f(trans[(k + 0) * TT + j] * LOG2EF);
            float e1 = __builtin_amdgcn_exp2f(trans[(k + 1) * TT + j] * LOG2EF);
            __bf16 h0 = (__bf16)e0, h1 = (__bf16)e1;
            ehi[blk][2 * v]     = h0;
            ehi[blk][2 * v + 1] = h1;
            elo[blk][2 * v]     = (__bf16)(e0 - (float)h0);
            elo[blk][2 * v + 1] = (__bf16)(e1 - (float)h1);
        }
    }

    // row-major remap used by init / phase-2 / final: 16 threads per batch row
    const int r2 = tid >> 4;        // batch row within tile
    const int lr = tid & 15;        // lane-in-row
    const int c0 = lr * 8;          // 8 contiguous tags per thread

    // ---- sequence length per row (mask[b][t] == (t < len_b)); int4 loads
    {
        const int4* mp = (const int4*)&mask[(size_t)(b0 + r2) * LL + lr * 32];
        int cnt = 0;
        #pragma unroll
        for (int i = 0; i < 8; ++i) {
            int4 m4 = mp[i];
            cnt += m4.x + m4.y + m4.z + m4.w;
        }
        #pragma unroll
        for (int d = 1; d < 16; d <<= 1) cnt += __shfl_xor(cnt, d, 32);
        if (lr == 0) s_len[r2] = cnt;
    }

    // ---- t = 0: alpha0 = start + logits[:,0,:]; rowmax + bf16 split of p
    {
        float a[8];
        const float4* ip = (const float4*)&inputs[(size_t)(b0 + r2) * LL * TT + c0];
        const float4* sp = (const float4*)&start_t[c0];
        float4 i0 = ip[0], i1 = ip[1], st0 = sp[0], st1 = sp[1];
        a[0] = st0.x + i0.x; a[1] = st0.y + i0.y; a[2] = st0.z + i0.z; a[3] = st0.w + i0.w;
        a[4] = st1.x + i1.x; a[5] = st1.y + i1.y; a[6] = st1.z + i1.z; a[7] = st1.w + i1.w;
        float4* ap = (float4*)&s_alpha[r2][c0];
        ap[0] = make_float4(a[0], a[1], a[2], a[3]);
        ap[1] = make_float4(a[4], a[5], a[6], a[7]);
        float mx = a[0];
        #pragma unroll
        for (int i = 1; i < 8; ++i) mx = fmaxf(mx, a[i]);
        #pragma unroll
        for (int d = 1; d < 16; d <<= 1) mx = fmaxf(mx, __shfl_xor(mx, d, 32));
        if (lr == 0) s_rowmax[r2] = mx;
        float e[8];
        #pragma unroll
        for (int i = 0; i < 8; ++i) e[i] = __builtin_amdgcn_exp2f((a[i] - mx) * LOG2EF);
        uint4 uh, ul;
        uh.x = bf16_pack(e[0], e[1]); uh.y = bf16_pack(e[2], e[3]);
        uh.z = bf16_pack(e[4], e[5]); uh.w = bf16_pack(e[6], e[7]);
        ul.x = bf16_pack(e[0] - bf16_hi_val(e[0]), e[1] - bf16_hi_val(e[1]));
        ul.y = bf16_pack(e[2] - bf16_hi_val(e[2]), e[3] - bf16_hi_val(e[3]));
        ul.z = bf16_pack(e[4] - bf16_hi_val(e[4]), e[5] - bf16_hi_val(e[5]));
        ul.w = bf16_pack(e[6] - bf16_hi_val(e[6]), e[7] - bf16_hi_val(e[7]));
        *(uint4*)&s_pah[r2][c0] = uh;
        *(uint4*)&s_pal[r2][c0] = ul;
    }
    __syncthreads();

    // per-lane register copy of the 8 row lengths this lane's D elements use
    int lenr[8];
    #pragma unroll
    for (int k = 0; k < 8; ++k) lenr[k] = s_len[k + (hi16 ? 8 : 0)];

    for (int t = 1; t < LL; ++t) {
        // ---- phase 1: s_j = (p_hi+p_lo) @ (E_hi+E_lo), K=128 in 4 blocks,
        // 3 independent accumulator chains (hides WMMA RAW hazard NOPs).
        v8f acc0 = {0, 0, 0, 0, 0, 0, 0, 0};
        v8f acc1 = {0, 0, 0, 0, 0, 0, 0, 0};
        v8f acc2 = {0, 0, 0, 0, 0, 0, 0, 0};
        const int ar = jn; // A-frag row = lane&15 in both halves
        #pragma unroll
        for (int blk = 0; blk < 4; ++blk) {
            // A frag in this code's contiguous k order: lanes<16: K 0..15; lanes>=16: K 16..31 (matches B)
            AFrag ah, al;
            int k0 = blk * 32 + (hi16 ? 16 : 0);
            int k1 = k0 + 8;
            ah.q[0] = *(const uint4*)&s_pah[ar][k0];
            ah.q[1] = *(const uint4*)&s_pah[ar][k1];
            al.q[0] = *(const uint4*)&s_pal[ar][k0];
            al.q[1] = *(const uint4*)&s_pal[ar][k1];
            acc0 = __builtin_amdgcn_wmma_f32_16x16x32_bf16(false, ah.v, false, ehi[blk],
                                                           (short)0, acc0, false, false);
            acc1 = __builtin_amdgcn_wmma_f32_16x16x32_bf16(false, ah.v, false, elo[blk],
                                                           (short)0, acc1, false, false);
            acc2 = __builtin_amdgcn_wmma_f32_16x16x32_bf16(false, al.v, false, ehi[blk],
                                                           (short)0, acc2, false, false);
        }
        // D layout: VGPR k -> row k (lanes 0-15) / row k+8 (lanes 16-31), col jn
        const float* ebase = &inputs[((size_t)(b0 + (hi16 ? 8 : 0)) * LL + t) * TT + j];
        #pragma unroll
        for (int k = 0; k < 8; ++k) {
            int row = k + (hi16 ? 8 : 0);
            float m = s_rowmax[row];
            const float* ep = ebase + (size_t)k * (LL * TT);
            float emit = *ep;
            if (t + 1 < LL) __builtin_prefetch(ep + TT, 0, 1); // next step's emissions
            float s  = acc0[k] + acc1[k] + acc2[k];
            float nv = m + __builtin_amdgcn_logf(s) * LN2F + emit;
            float ov = s_alpha[row][j];
            s_alpha[row][j] = (t < lenr[k]) ? nv : ov;
        }
        __syncthreads();

        // ---- phase 2: rowmax via shuffle-xor + exp/bf16-split of p (vectorized LDS)
        {
            const float4* ap = (const float4*)&s_alpha[r2][c0];
            float4 a0 = ap[0], a1 = ap[1];
            float a[8] = {a0.x, a0.y, a0.z, a0.w, a1.x, a1.y, a1.z, a1.w};
            float mx = a[0];
            #pragma unroll
            for (int i = 1; i < 8; ++i) mx = fmaxf(mx, a[i]);
            #pragma unroll
            for (int d = 1; d < 16; d <<= 1) mx = fmaxf(mx, __shfl_xor(mx, d, 32));
            if (lr == 0) s_rowmax[r2] = mx;
            float e[8];
            #pragma unroll
            for (int i = 0; i < 8; ++i) e[i] = __builtin_amdgcn_exp2f((a[i] - mx) * LOG2EF);
            uint4 uh, ul;
            uh.x = bf16_pack(e[0], e[1]); uh.y = bf16_pack(e[2], e[3]);
            uh.z = bf16_pack(e[4], e[5]); uh.w = bf16_pack(e[6], e[7]);
            ul.x = bf16_pack(e[0] - bf16_hi_val(e[0]), e[1] - bf16_hi_val(e[1]));
            ul.y = bf16_pack(e[2] - bf16_hi_val(e[2]), e[3] - bf16_hi_val(e[3]));
            ul.z = bf16_pack(e[4] - bf16_hi_val(e[4]), e[5] - bf16_hi_val(e[5]));
            ul.w = bf16_pack(e[6] - bf16_hi_val(e[6]), e[7] - bf16_hi_val(e[7]));
            *(uint4*)&s_pah[r2][c0] = uh;
            *(uint4*)&s_pal[r2][c0] = ul;
        }
        __syncthreads();
    }

    // ---- final: den[b] = logsumexp_j(alpha_final + end_t)
    {
        const float4* ap = (const float4*)&s_alpha[r2][c0];
        const float4* tp = (const float4*)&end_t[c0];
        float4 a0 = ap[0], a1 = ap[1], t0 = tp[0], t1 = tp[1];
        float a[8] = {a0.x + t0.x, a0.y + t0.y, a0.z + t0.z, a0.w + t0.w,
                      a1.x + t1.x, a1.y + t1.y, a1.z + t1.z, a1.w + t1.w};
        float mx = a[0];
        #pragma unroll
        for (int i = 1; i < 8; ++i) mx = fmaxf(mx, a[i]);
        #pragma unroll
        for (int d = 1; d < 16; d <<= 1) mx = fmaxf(mx, __shfl_xor(mx, d, 32));
        float sm = 0.f;
        #pragma unroll
        for (int i = 0; i < 8; ++i) sm += __builtin_amdgcn_exp2f((a[i] - mx) * LOG2EF);
        #pragma unroll
        for (int d = 1; d < 16; d <<= 1) sm += __shfl_xor(sm, d, 32);
        if (lr == 0) den[b0 + r2] = mx + __builtin_amdgcn_logf(sm) * LN2F;
    }
}

__global__ void __launch_bounds__(128)
crf_num_kernel(const float* __restrict__ inputs,
               const float* __restrict__ trans,
               const float* __restrict__ start_t,
               const float* __restrict__ end_t,
               const int*   __restrict__ tags,
               const int*   __restrict__ mask,
               float* __restrict__ num)
{
    __shared__ float sacc[128];
    __shared__ int   scnt[128];
    const int b = blockIdx.x, tid = threadIdx.x;
    float acc = 0.f;
    int   cnt = 0;
    for (int t = tid; t < LL; t += 128) {
        int mk = mask[b * LL + t];
        cnt += mk;
        if (mk) {
            int tg = tags[b * LL + t];
            if (t >= 1)      acc += trans[tags[b * LL + t - 1] * TT + tg];
            if (t <= LL - 2) acc += inputs[((size_t)b * LL + t) * TT + tg];
        }
    }
    sacc[tid] = acc; scnt[tid] = cnt;
    __syncthreads();
    for (int s = 64; s > 0; s >>= 1) {
        if (tid < s) { sacc[tid] += sacc[tid + s]; scnt[tid] += scnt[tid + s]; }
        __syncthreads();
    }
    if (tid == 0) {
        int last = tags[b * LL + (scnt[0] - 1)];
        float sc = sacc[0] + start_t[tags[b * LL]] + end_t[last];
        if (mask[b * LL + LL - 1])
            sc += inputs[((size_t)b * LL + LL - 1) * TT + last];
        num[b] = sc;
    }
}

__global__ void __launch_bounds__(256)
crf_reduce_kernel(const float* __restrict__ num,
                  const float* __restrict__ den,
                  float* __restrict__ out)
{
    __shared__ float s[256];
    const int tid = threadIdx.x;
    s[tid] = num[tid] - den[tid];
    __syncthreads();
    for (int k = 128; k > 0; k >>= 1) {
        if (tid < k) s[tid] += s[tid + k];
        __syncthreads();
    }
    if (tid == 0) out[0] = s[0];
}

extern "C" void kernel_launch(void* const* d_in, const int* in_sizes, int n_in,
                              void* d_out, int out_size, void* d_ws, size_t ws_size,
                              hipStream_t stream) {
    const float* inputs  = (const float*)d_in[0];
    const float* trans   = (const float*)d_in[1];
    const float* start_t = (const float*)d_in[2];
    const float* end_t   = (const float*)d_in[3];
    const int*   tags    = (const int*)d_in[4];
    const int*   mask    = (const int*)d_in[5];
    float* out = (float*)d_out;

    float* den = (float*)d_ws;       // [BB]
    float* num = den + BB;           // [BB]

    crf_forward_kernel<<<BB / 16, 256, 0, stream>>>(inputs, trans, start_t, end_t, mask, den);
    crf_num_kernel<<<BB, 128, 0, stream>>>(inputs, trans, start_t, end_t, tags, mask, num);
    crf_reduce_kernel<<<1, 256, 0, stream>>>(num, den, out);
}
